// AttentionCell_29403346109039
// MI455X (gfx1250) — matrix operand order repaired and checked by reading, not compiled, not run on hardware
//
#include <hip/hip_runtime.h>

#define B_   256
#define S_   256
#define DIN  512
#define DH   256
#define DOUT 128
#define TOUT 32
#define CONC 896   // DOUT + DH + DIN
#define NRZY 640   // DH + DH + DOUT
#define APAD 904   // CONC padded (+8 shorts) -> 452 dwords/row, 452 % 64 == 4 (bank-conflict-free)

typedef __bf16 bf16_t;
typedef __attribute__((ext_vector_type(16))) bf16_t v16bf;
typedef __attribute__((ext_vector_type(8)))  float  v8f;

// ---------- scalar helpers ----------
static __device__ __forceinline__ unsigned short f2bs(float f) {
  union { float f; unsigned u; } v; v.f = f;
  unsigned u = v.u;
  unsigned r = u + 0x7FFFu + ((u >> 16) & 1u);   // RNE
  return (unsigned short)(r >> 16);
}
static __device__ __forceinline__ float bs2f(unsigned short h) {
  union { unsigned u; float f; } v; v.u = ((unsigned)h) << 16; return v.f;
}
static __device__ __forceinline__ float fast_tanh(float x) {
#if __has_builtin(__builtin_amdgcn_tanhf)
  return __builtin_amdgcn_tanhf(x);      // native V_TANH_F32 (CDNA5 trans op)
#else
  return tanhf(x);
#endif
}
static __device__ __forceinline__ float fast_sigmoid(float x) {
  return 1.f / (1.f + __expf(-x));
}

union FragBF { v16bf v; unsigned short s[16]; uint4 q[2]; };

// ---------------------------------------------------------------------------
// Per-wave WMMA tile: 16 rows x 64 cols, K in steps of 32 (bf16 x bf16 -> f32).
// A: row-major MxK, f32 (converted on load) or bf16; works for global or LDS A.
// Bt: bf16 weights pre-transposed to [n][k], ldb = K.
// A frag: lane<16 -> row=lane, K kb..kb+7 & kb+16..kb+23 (kb=kk or kk+8).
// B frag: lane<16 -> col=lane&15, K kb..kb+7 & kb+16..kb+23 (same kb as A).
// C/D:    N = lane&15, M = (lane>=16 ? 8 : 0) + vgpr index.
// ---------------------------------------------------------------------------
template <bool AF32>
static __device__ __forceinline__ void
gemm_core_16x64(const void* __restrict__ A, long lda,
                const unsigned short* __restrict__ Bt, int ldb, int K,
                int row0, int col0, v8f acc[4])
{
  const int lane = threadIdx.x & 31;
  const int r    = lane & 15;
  const int hsel = lane >> 4;
  for (int kk = 0; kk < K; kk += 32) {
    FragBF a;
    const int kb = kk + (hsel ? 8 : 0);
    if (AF32) {
      const float* Arow = (const float*)A + (long)(row0 + r) * lda + kb;
      float4 lo0 = *(const float4*)(Arow + 0);
      float4 lo1 = *(const float4*)(Arow + 4);
      float4 hi0 = *(const float4*)(Arow + 16);
      float4 hi1 = *(const float4*)(Arow + 20);
      a.s[0]  = f2bs(lo0.x); a.s[1]  = f2bs(lo0.y); a.s[2]  = f2bs(lo0.z); a.s[3]  = f2bs(lo0.w);
      a.s[4]  = f2bs(lo1.x); a.s[5]  = f2bs(lo1.y); a.s[6]  = f2bs(lo1.z); a.s[7]  = f2bs(lo1.w);
      a.s[8]  = f2bs(hi0.x); a.s[9]  = f2bs(hi0.y); a.s[10] = f2bs(hi0.z); a.s[11] = f2bs(hi0.w);
      a.s[12] = f2bs(hi1.x); a.s[13] = f2bs(hi1.y); a.s[14] = f2bs(hi1.z); a.s[15] = f2bs(hi1.w);
    } else {
      const unsigned short* Arow = (const unsigned short*)A + (long)(row0 + r) * lda + kb;
      a.q[0] = *(const uint4*)(Arow);
      a.q[1] = *(const uint4*)(Arow + 16);
    }
#pragma unroll
    for (int j = 0; j < 4; ++j) {
      FragBF b;
      const unsigned short* Brow = Bt + (long)(col0 + 16 * j + r) * ldb + kk + hsel * 8;
      b.q[0] = *(const uint4*)(Brow);
      b.q[1] = *(const uint4*)(Brow + 16);
      acc[j] = __builtin_amdgcn_wmma_f32_16x16x32_bf16(
          false, a.v, false, b.v, (short)0, acc[j], false, false);
    }
  }
}

// ---------------- setup GEMMs ----------------

// wh = h @ W_ha  (bf16 output, (B*S) x DH); A is f32 h or bf16 h copy
template <bool AF32>
__global__ __launch_bounds__(32) void k_gemm_wh(
    const void* __restrict__ hsrc, const unsigned short* __restrict__ WhaT,
    unsigned short* __restrict__ wh)
{
  const int row0 = blockIdx.x * 16, col0 = blockIdx.y * 64;
  v8f zero = {}; v8f acc[4] = {zero, zero, zero, zero};
  gemm_core_16x64<AF32>(hsrc, DIN, WhaT, DIN, DIN, row0, col0, acc);
  const int lane = threadIdx.x & 31, r = lane & 15, hsel = lane >> 4;
#pragma unroll
  for (int j = 0; j < 4; ++j)
#pragma unroll
    for (int i = 0; i < 8; ++i)
      wh[(long)(row0 + hsel * 8 + i) * DH + col0 + 16 * j + r] = f2bs(acc[j][i]);
}

// s0 = tanh(h[:,0,:] @ W_init + b_init)
template <bool AF32>
__global__ __launch_bounds__(32) void k_gemm_s0(
    const void* __restrict__ hsrc, const unsigned short* __restrict__ WinitT,
    const float* __restrict__ b_init, float* __restrict__ sbuf)
{
  const int row0 = blockIdx.x * 16, col0 = blockIdx.y * 64;
  v8f zero = {}; v8f acc[4] = {zero, zero, zero, zero};
  gemm_core_16x64<AF32>(hsrc, (long)S_ * DIN, WinitT, DIN, DIN, row0, col0, acc);
  const int lane = threadIdx.x & 31, r = lane & 15, hsel = lane >> 4;
#pragma unroll
  for (int j = 0; j < 4; ++j)
#pragma unroll
    for (int i = 0; i < 8; ++i) {
      int m = row0 + hsel * 8 + i, n = col0 + 16 * j + r;
      sbuf[(long)m * DH + n] = fast_tanh(acc[j][i] + b_init[n]);
    }
}

// ---------------- fused per-step kernel 1: sW + attention + context + pack ----
// One block per batch row, 256 threads = 8 waves. HBF: h stored as bf16 copy.
template <bool HBF>
__global__ __launch_bounds__(256) void k_attn(
    const void* __restrict__ hsrc, const unsigned short* __restrict__ wh,
    const unsigned short* __restrict__ WsaT, const float* __restrict__ b_sa,
    const float* __restrict__ w_a, const float* __restrict__ sbuf,
    const float* __restrict__ ybuf, unsigned short* __restrict__ ysc)
{
  __shared__ float sh_s[DH];
  __shared__ float sh_sw[DH];
  __shared__ float sh_wa[DH];
  __shared__ float sh_e[S_];
  __shared__ float red[8];
  const int tid = threadIdx.x, b = blockIdx.x;
  const int wave = tid >> 5, lane = tid & 31;

  sh_s[tid]  = sbuf[b * DH + tid];
  sh_wa[tid] = w_a[tid];
  __syncthreads();

  // sW[n] = b_sa[n] + sum_k s[k] * W_sa[k][n]
  // wave-cooperative like the e-phase: wave w handles n = 32w..32w+31,
  // lanes split K contiguously (contiguous 512B per wave) + shuffle reduce.
  for (int n = wave * 32; n < wave * 32 + 32; ++n) {
    uint4 q = *(const uint4*)(WsaT + (long)n * DH + lane * 8);
    const unsigned short* w8 = (const unsigned short*)&q;
    float acc = 0.f;
#pragma unroll
    for (int i = 0; i < 8; ++i) acc += sh_s[lane * 8 + i] * bs2f(w8[i]);
#pragma unroll
    for (int off = 16; off >= 1; off >>= 1) acc += __shfl_xor(acc, off, 32);
    if (lane == 0) sh_sw[n] = acc + b_sa[n];
  }
  __syncthreads();

  // e[s] = sum_h tanh(wh[b,s,h] + sW[h]) * w_a[h]   (coalesced per wave)
  const unsigned short* whb = wh + (long)b * S_ * DH;
  for (int s = wave * 32; s < wave * 32 + 32; ++s) {
    uint4 q = *(const uint4*)(whb + (long)s * DH + lane * 8);
    const unsigned short* e8 = (const unsigned short*)&q;
    float acc = 0.f;
#pragma unroll
    for (int i = 0; i < 8; ++i)
      acc += fast_tanh(bs2f(e8[i]) + sh_sw[lane * 8 + i]) * sh_wa[lane * 8 + i];
#pragma unroll
    for (int off = 16; off >= 1; off >>= 1) acc += __shfl_xor(acc, off, 32);
    if (lane == 0) sh_e[s] = acc;
  }
  __syncthreads();

  // softmax over S
  float m = sh_e[tid];
#pragma unroll
  for (int off = 16; off >= 1; off >>= 1) m = fmaxf(m, __shfl_xor(m, off, 32));
  if (lane == 0) red[wave] = m;
  __syncthreads();
  if (tid == 0) { float x = red[0]; for (int i = 1; i < 8; ++i) x = fmaxf(x, red[i]); red[0] = x; }
  __syncthreads();
  m = red[0];
  __syncthreads();
  float p = __expf(sh_e[tid] - m);
  sh_e[tid] = p;
  float ssum = p;
#pragma unroll
  for (int off = 16; off >= 1; off >>= 1) ssum += __shfl_xor(ssum, off, 32);
  if (lane == 0) red[wave] = ssum;
  __syncthreads();
  if (tid == 0) { float x = 0.f; for (int i = 0; i < 8; ++i) x += red[i]; red[0] = x; }
  __syncthreads();
  const float inv = 1.f / red[0];

  // context: c[2*tid], c[2*tid+1]  (coalesced across threads at each s)
  float c0 = 0.f, c1 = 0.f;
  if (HBF) {
    const unsigned short* hb = (const unsigned short*)hsrc + (long)b * S_ * DIN + 2 * tid;
    for (int s = 0; s < S_; ++s) {
      float a = sh_e[s] * inv;
      unsigned u = *(const unsigned*)(hb + (long)s * DIN);
      c0 += a * bs2f((unsigned short)u);
      c1 += a * bs2f((unsigned short)(u >> 16));
    }
  } else {
    const float* hb = (const float*)hsrc + (long)b * S_ * DIN + 2 * tid;
    for (int s = 0; s < S_; ++s) {
      float a = sh_e[s] * inv;
      float2 v = *(const float2*)(hb + (long)s * DIN);
      c0 += a * v.x;
      c1 += a * v.y;
    }
  }

  // pack bf16 concat ysc = [y, s, c]
  const long base = (long)b * CONC;
  ysc[base + DOUT + DH + 2 * tid]     = f2bs(c0);
  ysc[base + DOUT + DH + 2 * tid + 1] = f2bs(c1);
  ysc[base + DOUT + tid] = f2bs(sh_s[tid]);
  if (tid < DOUT) ysc[base + tid] = f2bs(ybuf[b * DOUT + tid]);
}

// ---------------- fused per-step kernel 2: rzy GEMM + GRU + y softmax --------
// One block per 16 batch rows, 320 threads = 10 waves.
// Phase 1: G = ysc @ [W_r|W_z|W_y] (10 column tiles, one per wave), sigmoid r/z.
//          r*s overwrites the s-segment of the LDS A tile (ysc -> yrsc in place).
// Phase 2: waves 0-3: sp = tanh(yrsc @ W_sp + b_sp); s = (1-z)s + z*sp
//          waves 4-9: row softmax of y logits -> ybuf and d_out.
__global__ __launch_bounds__(320) void k_gates(
    const unsigned short* __restrict__ ysc,
    const unsigned short* __restrict__ WrzyT,
    const unsigned short* __restrict__ WspT,
    const float* __restrict__ b_r, const float* __restrict__ b_z,
    const float* __restrict__ b_y, const float* __restrict__ b_sp,
    float* __restrict__ sbuf, float* __restrict__ ybuf,
    float* __restrict__ out, int t)
{
  __shared__ unsigned short shA[16 * APAD];   // ysc rows (becomes yrsc in place)
  __shared__ float shZ[16 * DH];              // z gate
  __shared__ float shY[16 * DOUT];            // y logits
  const int tid  = threadIdx.x;
  const int wave = tid >> 5, lane = tid & 31;
  const int r = lane & 15, hsel = lane >> 4;
  const int row0 = blockIdx.x * 16;

  // stage A rows into LDS (16B vectors, padded rows)
  for (int idx = tid; idx < 16 * (CONC / 8); idx += 320) {
    int row = idx / (CONC / 8), c8 = idx % (CONC / 8);
    *(uint4*)&shA[row * APAD + c8 * 8] =
        *(const uint4*)&ysc[(long)(row0 + row) * CONC + c8 * 8];
  }
  __syncthreads();

  // Phase 1: 10 waves x (16x64) tiles over N = 640
  {
    v8f zero = {}; v8f acc[4] = {zero, zero, zero, zero};
    gemm_core_16x64<false>(shA, APAD, WrzyT, CONC, CONC, 0, wave * 64, acc);
#pragma unroll
    for (int j = 0; j < 4; ++j)
#pragma unroll
      for (int i = 0; i < 8; ++i) {
        int lrow = hsel * 8 + i;
        int n = wave * 64 + 16 * j + r;
        float v = acc[j][i];
        if (n < DH) {                           // r gate -> r*s into LDS A tile
          float g = fast_sigmoid(v + b_r[n]);
          shA[lrow * APAD + DOUT + n] = f2bs(g * sbuf[(long)(row0 + lrow) * DH + n]);
        } else if (n < 2 * DH) {                // z gate
          shZ[lrow * DH + (n - DH)] = fast_sigmoid(v + b_z[n - DH]);
        } else {                                // y logits
          shY[lrow * DOUT + (n - 2 * DH)] = v + b_y[n - 2 * DH];
        }
      }
  }
  __syncthreads();

  if (wave < 4) {
    // Phase 2a: sp GEMM + GRU blend (N = 256 over 4 waves)
    v8f zero = {}; v8f acc[4] = {zero, zero, zero, zero};
    gemm_core_16x64<false>(shA, APAD, WspT, CONC, CONC, 0, wave * 64, acc);
#pragma unroll
    for (int j = 0; j < 4; ++j)
#pragma unroll
      for (int i = 0; i < 8; ++i) {
        int lrow = hsel * 8 + i;
        int n = wave * 64 + 16 * j + r;
        float z  = shZ[lrow * DH + n];
        float sp = fast_tanh(acc[j][i] + b_sp[n]);
        long  si = (long)(row0 + lrow) * DH + n;
        sbuf[si] = (1.f - z) * sbuf[si] + z * sp;
      }
  } else {
    // Phase 2b: y softmax per row (rows distributed over waves 4..9)
    for (int row = wave - 4; row < 16; row += 6) {
      const float* yl = &shY[row * DOUT];
      float v0 = yl[lane], v1 = yl[lane + 32], v2 = yl[lane + 64], v3 = yl[lane + 96];
      float m = fmaxf(fmaxf(v0, v1), fmaxf(v2, v3));
#pragma unroll
      for (int off = 16; off >= 1; off >>= 1) m = fmaxf(m, __shfl_xor(m, off, 32));
      float p0 = __expf(v0 - m), p1 = __expf(v1 - m),
            p2 = __expf(v2 - m), p3 = __expf(v3 - m);
      float ss = p0 + p1 + p2 + p3;
#pragma unroll
      for (int off = 16; off >= 1; off >>= 1) ss += __shfl_xor(ss, off, 32);
      float inv = 1.f / ss;
      int mrow = row0 + row;
      float* yb = ybuf + (long)mrow * DOUT;
      float* ob = out + ((long)mrow * TOUT + t) * DOUT;
      yb[lane] = p0 * inv; yb[lane + 32] = p1 * inv;
      yb[lane + 64] = p2 * inv; yb[lane + 96] = p3 * inv;
      ob[lane] = p0 * inv; ob[lane + 32] = p1 * inv;
      ob[lane + 64] = p2 * inv; ob[lane + 96] = p3 * inv;
    }
  }
}

// ---------------- conversion / init ----------------
// dst[n*K + k] = bf16(src[k*N + n])
__global__ void k_convt(unsigned short* __restrict__ dst,
                        const float* __restrict__ src, int K, int N)
{
  long total = (long)K * N;
  for (long idx = (long)blockIdx.x * blockDim.x + threadIdx.x; idx < total;
       idx += (long)gridDim.x * blockDim.x) {
    long n = idx / K, k = idx % K;
    dst[idx] = f2bs(src[k * (long)N + n]);
  }
}

// straight f32 -> bf16 copy, 2 elements per thread-iter (packed 32-bit stores)
__global__ void k_convbf(unsigned* __restrict__ dst, const float* __restrict__ src, long npairs)
{
  for (long i = (long)blockIdx.x * blockDim.x + threadIdx.x; i < npairs;
       i += (long)gridDim.x * blockDim.x) {
    float2 v = *(const float2*)(src + 2 * i);
    dst[i] = (unsigned)f2bs(v.x) | ((unsigned)f2bs(v.y) << 16);
  }
}

__global__ void k_zero(float* __restrict__ p, int n)
{
  int i = blockIdx.x * blockDim.x + threadIdx.x;
  if (i < n) p[i] = 0.f;
}

// ---------------------------------------------------------------------------
extern "C" void kernel_launch(void* const* d_in, const int* in_sizes, int n_in,
                              void* d_out, int out_size, void* d_ws, size_t ws_size,
                              hipStream_t stream)
{
  (void)in_sizes; (void)n_in; (void)out_size;
  const float* h      = (const float*)d_in[0];
  const float* W_ha   = (const float*)d_in[1];
  const float* W_sa   = (const float*)d_in[2];
  const float* b_sa   = (const float*)d_in[3];
  const float* w_a    = (const float*)d_in[4];
  const float* W_init = (const float*)d_in[5];
  const float* b_init = (const float*)d_in[6];
  const float* W_r    = (const float*)d_in[7];
  const float* b_r    = (const float*)d_in[8];
  const float* W_z    = (const float*)d_in[9];
  const float* b_z    = (const float*)d_in[10];
  const float* W_sp   = (const float*)d_in[11];
  const float* b_sp   = (const float*)d_in[12];
  const float* W_y    = (const float*)d_in[13];
  const float* b_y    = (const float*)d_in[14];
  float* out = (float*)d_out;

  char* wsp = (char*)d_ws;
  size_t off = 0;
  auto alloc = [&](size_t bytes) -> char* {
    char* p = wsp + off;
    off += (bytes + 255) & ~(size_t)255;
    return p;
  };

  unsigned short* wh     = (unsigned short*)alloc((size_t)B_ * S_ * DH * 2);  // 32 MB
  unsigned short* WhaT   = (unsigned short*)alloc((size_t)DH * DIN * 2);
  unsigned short* WinitT = (unsigned short*)alloc((size_t)DH * DIN * 2);
  unsigned short* WsaT   = (unsigned short*)alloc((size_t)DH * DH * 2);
  unsigned short* WrzyT  = (unsigned short*)alloc((size_t)NRZY * CONC * 2);
  unsigned short* WspT   = (unsigned short*)alloc((size_t)DH * CONC * 2);
  float*          sbuf   = (float*)alloc((size_t)B_ * DH * 4);
  float*          ybuf   = (float*)alloc((size_t)B_ * DOUT * 4);
  unsigned short* ysc    = (unsigned short*)alloc((size_t)B_ * CONC * 2);
  size_t base_need = off;
  const size_t hbf_bytes = (size_t)B_ * S_ * DIN * 2;                         // 64 MB
  const bool use_hbf = (ws_size >= base_need + hbf_bytes + 256);
  unsigned short* hbf = use_hbf ? (unsigned short*)alloc(hbf_bytes) : nullptr;

  const dim3 cb(256);
  // one-time weight transpose+convert to bf16 [n][k]
  k_convt<<<dim3(256), cb, 0, stream>>>(WhaT,   W_ha,   DIN,  DH);
  k_convt<<<dim3(256), cb, 0, stream>>>(WinitT, W_init, DIN,  DH);
  k_convt<<<dim3(64),  cb, 0, stream>>>(WsaT,   W_sa,   DH,   DH);
  k_convt<<<dim3(256), cb, 0, stream>>>(WrzyT,                         W_r, CONC, DH);
  k_convt<<<dim3(256), cb, 0, stream>>>(WrzyT + (size_t)DH * CONC,     W_z, CONC, DH);
  k_convt<<<dim3(128), cb, 0, stream>>>(WrzyT + (size_t)2 * DH * CONC, W_y, CONC, DOUT);
  k_convt<<<dim3(256), cb, 0, stream>>>(WspT,   W_sp,   CONC, DH);
  k_zero<<<dim3((B_ * DOUT + 255) / 256), cb, 0, stream>>>(ybuf, B_ * DOUT);

  const dim3 gwh((B_ * S_) / 16, DH / 64), gsm(B_ / 16, DH / 64);
  if (use_hbf) {
    k_convbf<<<dim3(2048), cb, 0, stream>>>((unsigned*)hbf, h, (long)B_ * S_ * DIN / 2);
    k_gemm_wh<false><<<gwh, dim3(32), 0, stream>>>(hbf, WhaT, wh);
    k_gemm_s0<false><<<gsm, dim3(32), 0, stream>>>(hbf, WinitT, b_init, sbuf);
  } else {
    k_gemm_wh<true><<<gwh, dim3(32), 0, stream>>>(h, WhaT, wh);
    k_gemm_s0<true><<<gsm, dim3(32), 0, stream>>>(h, WinitT, b_init, sbuf);
  }

  for (int t = 0; t < TOUT; ++t) {
    if (use_hbf)
      k_attn<true><<<dim3(B_), dim3(256), 0, stream>>>(hbf, wh, WsaT, b_sa, w_a,
                                                       sbuf, ybuf, ysc);
    else
      k_attn<false><<<dim3(B_), dim3(256), 0, stream>>>(h, wh, WsaT, b_sa, w_a,
                                                        sbuf, ybuf, ysc);
    k_gates<<<dim3(B_ / 16), dim3(320), 0, stream>>>(ysc, WrzyT, WspT,
                                                     b_r, b_z, b_y, b_sp,
                                                     sbuf, ybuf, out, t);
  }
}
